// AdditiveAttention_12953621365263
// MI455X (gfx1250) — matrix operand order repaired and checked by reading, not compiled, not run on hardware
//
#include <hip/hip_runtime.h>
#include <math.h>

// Problem constants (from reference): B=32, S=4096, E=512, D=512, A=256
#define Bb 32
#define Ss 4096
#define Ee 512
#define Dd 512
#define Aa 256

#define ROWS_PER_BLOCK 128     // 8 row-groups x 16 rows; 16 waves = 8 rowgrp x 2 nhalf
#define THREADS 512            // 16 waves of 32
#define LDS_STRIDE 520         // bf16 elems per W_enc row (512 + 8 pad -> bank-conflict-free b128)
#define LDS_BYTES (Aa * LDS_STRIDE * 2)   // 266240 B (+1KB static) <= 320KB WGP LDS

typedef __attribute__((ext_vector_type(16))) __bf16 v16bf;
typedef __attribute__((ext_vector_type(8)))  float  v8f;

__device__ __forceinline__ unsigned short f32_to_bf16(float f) {
    unsigned int u = __float_as_uint(f);
    u += 0x7FFFu + ((u >> 16) & 1u);   // round-to-nearest-even
    return (unsigned short)(u >> 16);
}

// ---------------------------------------------------------------------------
// One-shot: convert W_enc (f32, [A][E]) -> bf16 in workspace. 8 elems/thread.
// ---------------------------------------------------------------------------
__global__ void wprep_kernel(const float* __restrict__ Wenc,
                             unsigned short* __restrict__ wbf) {
    int i = (blockIdx.x * 256 + threadIdx.x) * 8;
    float4 f0 = *(const float4*)(Wenc + i);
    float4 f1 = *(const float4*)(Wenc + i + 4);
    union { uint4 q; unsigned short u[8]; } o;
    o.u[0] = f32_to_bf16(f0.x); o.u[1] = f32_to_bf16(f0.y);
    o.u[2] = f32_to_bf16(f0.z); o.u[3] = f32_to_bf16(f0.w);
    o.u[4] = f32_to_bf16(f1.x); o.u[5] = f32_to_bf16(f1.y);
    o.u[6] = f32_to_bf16(f1.z); o.u[7] = f32_to_bf16(f1.w);
    *(uint4*)&wbf[i] = o.q;
}

// ---------------------------------------------------------------------------
// s_i[b][a] = sum_d decoder_hidden[b][d] * W_dec[a][d]   (tiny: 32x256 dots)
// ---------------------------------------------------------------------------
__global__ void si_kernel(const float* __restrict__ dh,
                          const float* __restrict__ Wdec,
                          float* __restrict__ s_ws) {
    int b = blockIdx.x;
    int a = threadIdx.x;
    const float* x = dh + b * Dd;
    const float* w = Wdec + a * Dd;
    float acc = 0.f;
    for (int d = 0; d < Dd; d += 4) {
        float4 xv = *(const float4*)(x + d);
        float4 wv = *(const float4*)(w + d);
        acc += xv.x * wv.x + xv.y * wv.y + xv.z * wv.z + xv.w * wv.w;
    }
    s_ws[b * Aa + a] = acc;
}

// ---------------------------------------------------------------------------
// Main GEMM (BF16 WMMA): h[row][n] = sum_e enc[row][e] * W_enc[n][e]
//   PASS 0: energy[row] = b_attn + sum_n tanh(h + s_i[b][n]) * W_attn[n]
//   PASS 1: out[row][n] = alpha[row] * h[row][n]
// Wave layout: wave = rowgrp*2 + nhalf; each wave does 16 rows x 128 cols
// (8 accumulator tiles = 64 VGPRs -> no spills). Pre-converted bf16 W_enc is
// staged into LDS with raw uint4 copies (no conversion VALU in hot kernels).
// ---------------------------------------------------------------------------
template <int PASS>
__global__ void __launch_bounds__(THREADS, 1)
attn_gemm_kernel(const float* __restrict__ enc,
                 const unsigned short* __restrict__ wbf,
                 const float* __restrict__ Wattn,
                 const float* __restrict__ battn,
                 const float* __restrict__ s_ws,
                 float* __restrict__ energy_ws,
                 const float* __restrict__ alpha_ws,
                 float* __restrict__ out) {
    extern __shared__ unsigned short ldsW[];   // bf16 bits, [Aa][LDS_STRIDE]
    __shared__ float epart[16][16];            // per-wave partial energy, 1KB

    const int tid = threadIdx.x;

    // Stage pre-converted bf16 W_enc into padded LDS: 32 x b128 per thread.
    for (int i = tid; i < (Aa * Ee) / 8; i += THREADS) {
        int idx = i * 8;
        int a = idx >> 9;        // / 512
        int e = idx & 511;
        *(uint4*)&ldsW[a * LDS_STRIDE + e] = *(const uint4*)&wbf[idx];
    }
    __syncthreads();

    const int wave   = tid >> 5;
    const int lane   = tid & 31;
    const int half   = lane >> 4;     // 0: lanes 0-15, 1: lanes 16-31
    const int ln     = lane & 15;
    const int rowgrp = wave >> 1;
    const int nhalf  = wave & 1;      // which 128-column half of N this wave owns

    const int rowBase = blockIdx.x * ROWS_PER_BLOCK + rowgrp * 16;
    const int aRow    = rowBase + ln;            // A-matrix row for this lane
    const float* arow = enc + (size_t)aRow * Ee;

    v8f acc[8];
    const v8f zeroacc = {};
    #pragma unroll
    for (int t = 0; t < 8; ++t) acc[t] = zeroacc;

    for (int kk = 0; kk < 16; ++kk) {           // K = 512, 32 per step
        // A fragment, 16x32 bf16: lane ln = row; half selects K 0-15/16-31
        const int c0 = kk * 32 + half * 16;
        float4 a0 = *(const float4*)(arow + c0);
        float4 a1 = *(const float4*)(arow + c0 + 4);
        float4 a2 = *(const float4*)(arow + c0 + 8);
        float4 a3 = *(const float4*)(arow + c0 + 12);
        union { v16bf v; unsigned short u[16]; } af;
        af.u[0]  = f32_to_bf16(a0.x); af.u[1]  = f32_to_bf16(a0.y);
        af.u[2]  = f32_to_bf16(a0.z); af.u[3]  = f32_to_bf16(a0.w);
        af.u[4]  = f32_to_bf16(a1.x); af.u[5]  = f32_to_bf16(a1.y);
        af.u[6]  = f32_to_bf16(a1.z); af.u[7]  = f32_to_bf16(a1.w);
        af.u[8]  = f32_to_bf16(a2.x); af.u[9]  = f32_to_bf16(a2.y);
        af.u[10] = f32_to_bf16(a2.z); af.u[11] = f32_to_bf16(a2.w);
        af.u[12] = f32_to_bf16(a3.x); af.u[13] = f32_to_bf16(a3.y);
        af.u[14] = f32_to_bf16(a3.z); af.u[15] = f32_to_bf16(a3.w);

        #pragma unroll
        for (int t = 0; t < 8; ++t) {
            // B fragment, 32x16: lane ln = column n, half selects K 0-15/16-31
            const int n = nhalf * 128 + t * 16 + ln;
            union { v16bf v; uint4 q[2]; } bf;
            const uint4* bp =
                (const uint4*)&ldsW[n * LDS_STRIDE + kk * 32 + half * 16];
            bf.q[0] = bp[0];
            bf.q[1] = bp[1];
            acc[t] = __builtin_amdgcn_wmma_f32_16x16x32_bf16(
                false, af.v, false, bf.v, (short)0, acc[t], false, false);
        }
    }

    // Epilogues. C/D layout: VGPR v -> rows v (lanes 0-15) / v+8 (lanes 16-31),
    // column n = nhalf*128 + t*16 + ln.
    if (PASS == 0) {
        const int b = blockIdx.x >> 5;          // 32 blocks per batch (4096/128)
        float eacc[8];
        #pragma unroll
        for (int v = 0; v < 8; ++v) eacc[v] = 0.f;
        #pragma unroll
        for (int t = 0; t < 8; ++t) {
            int n = nhalf * 128 + t * 16 + ln;
            float sv = s_ws[b * Aa + n];
            float wv = Wattn[n];
            #pragma unroll
            for (int v = 0; v < 8; ++v)
                eacc[v] += tanhf(acc[t][v] + sv) * wv;
        }
        // reduce across the 16 lanes of each half-wave
        #pragma unroll
        for (int m = 1; m < 16; m <<= 1) {
            #pragma unroll
            for (int v = 0; v < 8; ++v)
                eacc[v] += __shfl_xor(eacc[v], m, 32);
        }
        if (ln == 0) {
            #pragma unroll
            for (int v = 0; v < 8; ++v)
                epart[wave][half * 8 + v] = eacc[v];
        }
        __syncthreads();
        // combine the two N-halves of each row-group
        if (nhalf == 0 && ln == 0) {
            float bb = battn[0];
            int r0 = rowBase + half * 8;
            #pragma unroll
            for (int v = 0; v < 8; ++v)
                energy_ws[r0 + v] =
                    epart[wave][half * 8 + v] + epart[wave + 1][half * 8 + v] + bb;
        }
    } else {
        int r0 = rowBase + half * 8;
        float al[8];
        #pragma unroll
        for (int v = 0; v < 8; ++v) al[v] = alpha_ws[r0 + v];
        #pragma unroll
        for (int t = 0; t < 8; ++t) {
            int n = nhalf * 128 + t * 16 + ln;
            #pragma unroll
            for (int v = 0; v < 8; ++v)
                out[(size_t)(r0 + v) * Aa + n] = al[v] * acc[t][v];
        }
    }
}

// ---------------------------------------------------------------------------
// Softmax over S per batch; writes alpha to ws (for pass 1) and to d_out tail.
// ---------------------------------------------------------------------------
__global__ void softmax_kernel(const float* __restrict__ energy,
                               float* __restrict__ alpha_ws,
                               float* __restrict__ alpha_out) {
    __shared__ float red[256];
    const int b = blockIdx.x, tid = threadIdx.x;
    const float* e = energy + b * Ss;

    float lmax = -3.0e38f;
    for (int i = tid; i < Ss; i += 256) lmax = fmaxf(lmax, e[i]);
    red[tid] = lmax; __syncthreads();
    for (int s = 128; s > 0; s >>= 1) {
        if (tid < s) red[tid] = fmaxf(red[tid], red[tid + s]);
        __syncthreads();
    }
    float mx = red[0]; __syncthreads();

    float lsum = 0.f;
    for (int i = tid; i < Ss; i += 256) lsum += expf(e[i] - mx);
    red[tid] = lsum; __syncthreads();
    for (int s = 128; s > 0; s >>= 1) {
        if (tid < s) red[tid] += red[tid + s];
        __syncthreads();
    }
    float inv = 1.0f / red[0];

    for (int i = tid; i < Ss; i += 256) {
        float a = expf(e[i] - mx) * inv;
        alpha_ws[b * Ss + i] = a;
        alpha_out[b * Ss + i] = a;
    }
}

// ---------------------------------------------------------------------------
extern "C" void kernel_launch(void* const* d_in, const int* in_sizes, int n_in,
                              void* d_out, int out_size, void* d_ws, size_t ws_size,
                              hipStream_t stream) {
    const float* enc   = (const float*)d_in[0];   // [B,S,E]
    const float* dh    = (const float*)d_in[1];   // [1,B,D]
    const float* Wenc  = (const float*)d_in[2];   // [A,E]
    const float* Wdec  = (const float*)d_in[3];   // [A,D]
    const float* Wattn = (const float*)d_in[4];   // [1,A]
    const float* battn = (const float*)d_in[5];   // [1]

    float* out       = (float*)d_out;                       // [B,S,A] then [B,S]
    float* alpha_out = out + (size_t)Bb * Ss * Aa;

    // workspace: s_i (8K f) | energy (128K f) | alpha (128K f) | wbf (256KB)
    float* s_ws      = (float*)d_ws;
    float* energy_ws = s_ws + Bb * Aa;
    float* alpha_ws  = energy_ws + Bb * Ss;
    unsigned short* wbf = (unsigned short*)(alpha_ws + Bb * Ss);

    wprep_kernel<<<(Aa * Ee) / (256 * 8), 256, 0, stream>>>(Wenc, wbf);
    si_kernel<<<Bb, Aa, 0, stream>>>(dh, Wdec, s_ws);

    dim3 grid((Bb * Ss) / ROWS_PER_BLOCK);   // 1024 blocks
    attn_gemm_kernel<0><<<grid, THREADS, LDS_BYTES, stream>>>(
        enc, wbf, Wattn, battn, s_ws, energy_ws, nullptr, nullptr);

    softmax_kernel<<<Bb, 256, 0, stream>>>(energy_ws, alpha_ws, alpha_out);

    attn_gemm_kernel<1><<<grid, THREADS, LDS_BYTES, stream>>>(
        enc, wbf, Wattn, battn, nullptr, nullptr, alpha_ws, out);
}
